// VectorQuantizerFixed_52261162057757
// MI455X (gfx1250) — matrix operand order repaired and checked by reading, not compiled, not run on hardware
//
#include <hip/hip_runtime.h>
#include <hip/hip_bf16.h>
#include <math.h>

#define BN_TOK 32768
#define DIMX   256
#define DC     1024
#define KCB    1024
#define LDT    40   // padded K-stride for LDS tiles (80B rows, 16B aligned)

typedef __attribute__((ext_vector_type(16))) __bf16 v16bf;
typedef __attribute__((ext_vector_type(8)))  float  v8f;
typedef int v4i_ __attribute__((vector_size(16)));   // matches builtin param type

// ---------- CDNA5 async global->LDS copy (ASYNCcnt path), with sync fallback ----------
#if __has_builtin(__builtin_amdgcn_global_load_async_to_lds_b128) && \
    __has_builtin(__builtin_amdgcn_s_wait_asynccnt)
#define ASYNC_LDS_COPY 1
#else
#define ASYNC_LDS_COPY 0
#endif

__device__ __forceinline__ void cp16_async(const unsigned short* __restrict__ g,
                                           unsigned short* l) {
#if ASYNC_LDS_COPY
  // generic->AS1 is value-identical; generic LDS addr low 32 bits == LDS offset (aperture rule)
  __builtin_amdgcn_global_load_async_to_lds_b128(
      (__attribute__((address_space(1))) v4i_*)(unsigned long long)g,
      (__attribute__((address_space(3))) v4i_*)(unsigned)(unsigned long long)l,
      0, 0);
#else
  *(uint4*)l = *(const uint4*)g;
#endif
}
__device__ __forceinline__ void wait_async() {
#if ASYNC_LDS_COPY
  __builtin_amdgcn_s_wait_asynccnt(0);
#endif
}

// ---------- bf16 helpers (bit-exact RNE, no __bf16 scalar arithmetic) ----------
__device__ __forceinline__ unsigned short f2bf(float f) {
  union { float f; unsigned u; } v; v.f = f;
  unsigned r = v.u + 0x7FFFu + ((v.u >> 16) & 1u);
  return (unsigned short)(r >> 16);
}
__device__ __forceinline__ float bf2f(unsigned short h) {
  union { unsigned u; float f; } v; v.u = ((unsigned)h) << 16;
  return v.f;
}
__device__ __forceinline__ v8f vzero8() { v8f z = {0.f,0.f,0.f,0.f,0.f,0.f,0.f,0.f}; return z; }

// ---------- WMMA fragment loaders (wave32, 16x16x32 bf16) ----------
// A (16xK row-major tile): lane<16 -> M=lane, K={0..7,16..23}; lane>=16 -> M=lane-16, K={8..15,24..31}
__device__ __forceinline__ v16bf load_frag_a(const unsigned short* s, int ld) {
  int lane = threadIdx.x & 31;
  int m  = lane & 15;
  int kb = (lane & 16) ? 8 : 0;
  const unsigned short* p = s + m * ld + kb;
  union { v16bf v; uint4 q[2]; } u;
  u.q[0] = *(const uint4*)(p);
  u.q[1] = *(const uint4*)(p + 16);
  return u.v;
}
// B stored [n][k] in LDS: lane<16 -> N=lane, K={0..7,16..23}; lane>=16 -> N=lane-16, K={8..15,24..31}
__device__ __forceinline__ v16bf load_frag_b(const unsigned short* s, int ld) {
  int lane = threadIdx.x & 31;
  int n  = lane & 15;
  int kb = (lane & 16) ? 8 : 0;
  const unsigned short* p = s + n * ld + kb;
  union { v16bf v; uint4 q[2]; } u;
  u.q[0] = *(const uint4*)(p);
  u.q[1] = *(const uint4*)(p + 16);
  return u.v;
}
__device__ __forceinline__ v8f wmma_bf16(v16bf a, v16bf b, v8f c) {
  return __builtin_amdgcn_wmma_f32_16x16x32_bf16(false, a, false, b, (short)0, c, false, false);
}

// ---------- workspace layout (bytes, all 16B aligned) ----------
#define WS_EMB    0u                               // bf16 embed    [1024][1024]   2 MB
#define WS_PREWT  (2u << 20)                       // bf16 pre_w^T  [1024][256]  512 KB
#define WS_POSTWT (WS_PREWT  + (512u << 10))       // bf16 post_w^T [256][1024]  512 KB
#define WS_ESQ    (WS_POSTWT + (512u << 10))       // f32  ||e||^2  [1024]         4 KB
#define WS_LOSS   (WS_ESQ    + 4096u)              // f32 accumulator            256 B
#define WS_IDX    (WS_LOSS   + 256u)               // i32 indices   [BN]         128 KB
#define WS_ACT    (WS_IDX    + (BN_TOK * 4u))      // bf16 gelu(ln(x)) [BN][256]  16 MB
#define WS_FLAT   (WS_ACT    + BN_TOK * DIMX * 2u) // bf16 flat     [BN][1024]    64 MB

// ================= K0: weight conversion / transpose, zero loss =================
__global__ void k_prep(const float* __restrict__ embed, const float* __restrict__ pre_w,
                       const float* __restrict__ post_w,
                       unsigned short* __restrict__ embB, unsigned short* __restrict__ preWT,
                       unsigned short* __restrict__ postWT, float* __restrict__ lossAcc) {
  int i = blockIdx.x * blockDim.x + threadIdx.x;
  int stride = gridDim.x * blockDim.x;
  if (i == 0) lossAcc[0] = 0.f;
  const int nEmb = KCB * DC, nPre = DIMX * DC, nPost = DC * DIMX;
  for (int t = i; t < nEmb + nPre + nPost; t += stride) {
    if (t < nEmb) {
      embB[t] = f2bf(embed[t]);
    } else if (t < nEmb + nPre) {
      int j = t - nEmb;            // pre_w (in=256, out=1024) row-major
      int k = j / DC, n = j % DC;
      preWT[n * DIMX + k] = f2bf(pre_w[j]);
    } else {
      int j = t - nEmb - nPre;     // post_w (in=1024, out=256) row-major
      int k = j / DIMX, n = j % DIMX;
      postWT[n * DC + k] = f2bf(post_w[j]);
    }
  }
}

// ================= K1: ||e_k||^2 (one wave per codebook row) =================
__global__ void k_esq(const float* __restrict__ embed, float* __restrict__ esq) {
  int warp = (blockIdx.x * blockDim.x + threadIdx.x) >> 5;
  int lane = threadIdx.x & 31;
  if (warp >= KCB) return;
  const float* row = embed + (size_t)warp * DC;
  float s = 0.f;
  for (int c = lane; c < DC; c += 32) { float v = row[c]; s += v * v; }
  #pragma unroll
  for (int off = 16; off; off >>= 1) s += __shfl_xor(s, off, 32);
  if (lane == 0) esq[warp] = s;
}

// ================= K2: LayerNorm + exact GELU -> bf16 (one wave per token) =================
__global__ void k_ln(const float* __restrict__ x, const float* __restrict__ g,
                     const float* __restrict__ b, unsigned short* __restrict__ actB) {
  int warp = threadIdx.x >> 5, lane = threadIdx.x & 31;
  int row = blockIdx.x * 8 + warp;
  const float* xr = x + (size_t)row * DIMX;
  float4 v0 = ((const float4*)xr)[lane * 2];
  float4 v1 = ((const float4*)xr)[lane * 2 + 1];
  float vals[8] = {v0.x, v0.y, v0.z, v0.w, v1.x, v1.y, v1.z, v1.w};
  float s = 0.f, ss = 0.f;
  #pragma unroll
  for (int i = 0; i < 8; i++) { s += vals[i]; ss += vals[i] * vals[i]; }
  #pragma unroll
  for (int off = 16; off; off >>= 1) { s += __shfl_xor(s, off, 32); ss += __shfl_xor(ss, off, 32); }
  float mean = s * (1.f / DIMX);
  float var  = ss * (1.f / DIMX) - mean * mean;
  float rs   = rsqrtf(var + 1e-5f);
  union { unsigned short h[8]; uint4 q; } o;
  #pragma unroll
  for (int i = 0; i < 8; i++) {
    int c = lane * 8 + i;
    float y = (vals[i] - mean) * rs * g[c] + b[c];
    float ge = 0.5f * y * (1.f + erff(y * 0.70710678118654752f));
    o.h[i] = f2bf(ge);
  }
  *(uint4*)(actB + (size_t)row * DIMX + lane * 8) = o.q;
}

// ================= K3: pre_proj GEMM (act[BN,256] x preW[256,1024]) -> flat bf16 =================
__global__ void k_gemm1(const unsigned short* __restrict__ actB,
                        const unsigned short* __restrict__ preWT,
                        const float* __restrict__ pre_b, unsigned short* __restrict__ flatB) {
  __shared__ __align__(16) unsigned short sA[64 * LDT];
  __shared__ __align__(16) unsigned short sB[128 * LDT];
  int bm = blockIdx.x >> 3, bn = blockIdx.x & 7;
  int row0 = bm * 64, col0 = bn * 128;
  int t = threadIdx.x, wave = t >> 5, lane = t & 31;
  int wm = wave >> 1, wn = wave & 1;
  v8f acc[4];
  #pragma unroll
  for (int j = 0; j < 4; j++) acc[j] = vzero8();
  for (int kk = 0; kk < DIMX; kk += 32) {
    __syncthreads();
    { int r = t >> 2, kq = (t & 3) * 8;
      cp16_async(actB + (size_t)(row0 + r) * DIMX + kk + kq, sA + r * LDT + kq); }
    #pragma unroll
    for (int i = 0; i < 2; i++) {
      int idx = t + i * 256, n = idx >> 2, kq = (idx & 3) * 8;
      cp16_async(preWT + (size_t)(col0 + n) * DIMX + kk + kq, sB + n * LDT + kq);
    }
    wait_async();
    __syncthreads();
    v16bf a = load_frag_a(sA + (wm * 16) * LDT, LDT);
    #pragma unroll
    for (int j = 0; j < 4; j++) {
      v16bf bf = load_frag_b(sB + (wn * 64 + j * 16) * LDT, LDT);
      acc[j] = wmma_bf16(a, bf, acc[j]);
    }
  }
  int mrow = (lane & 16) ? 8 : 0, ncol = lane & 15;
  #pragma unroll
  for (int j = 0; j < 4; j++) {
    int gc = col0 + wn * 64 + j * 16 + ncol;
    float pb = pre_b[gc];
    #pragma unroll
    for (int r = 0; r < 8; r++) {
      int gm = row0 + wm * 16 + mrow + r;
      flatB[(size_t)gm * DC + gc] = f2bf(acc[j][r] + pb);
    }
  }
}

// ================= K4: distance GEMM + argmin (64 tokens/block, 4 code chunks, =================
// =================     double-buffered async LDS pipeline, 1 barrier per k-step) ==============
__global__ void k_argmin(const unsigned short* __restrict__ flatB,
                         const unsigned short* __restrict__ embB,
                         const float* __restrict__ esq, int* __restrict__ wsIdx,
                         float* __restrict__ outIdx) {
  __shared__ __align__(16) unsigned short sA[2][64 * LDT];   // 10 KB
  __shared__ __align__(16) unsigned short sB[2][256 * LDT];  // 40 KB
  __shared__ float redV[64][8];
  __shared__ int   redN[64][8];
  int row0 = blockIdx.x * 64;
  int t = threadIdx.x, wave = t >> 5, lane = t & 31;
  int ar = t >> 2, akq = (t & 3) * 8;   // A-tile task: 64 rows x 4 quads
  float bestv[32]; int bestn[32];
  #pragma unroll
  for (int i = 0; i < 32; i++) { bestv[i] = 3.4e38f; bestn[i] = 0; }

  for (int chunk = 0; chunk < 4; chunk++) {
    int code0 = chunk * 256;
    v8f acc[4][2];
    #pragma unroll
    for (int mf = 0; mf < 4; mf++)
      #pragma unroll
      for (int nf = 0; nf < 2; nf++) acc[mf][nf] = vzero8();
    // prologue: fill buffer 0 with kk=0 tiles
    cp16_async(flatB + (size_t)(row0 + ar) * DC + akq, sA[0] + ar * LDT + akq);
    #pragma unroll
    for (int i = 0; i < 4; i++) {
      int idx = t + i * 256, n = idx >> 2, kq = (idx & 3) * 8;
      cp16_async(embB + (size_t)(code0 + n) * DC + kq, sB[0] + n * LDT + kq);
    }
    wait_async();
    __syncthreads();
    int cur = 0;
    for (int kk = 0; kk < DC; kk += 32) {
      int nxt = cur ^ 1;
      if (kk + 32 < DC) {   // prefetch next k-slice into the other buffer
        cp16_async(flatB + (size_t)(row0 + ar) * DC + kk + 32 + akq, sA[nxt] + ar * LDT + akq);
        #pragma unroll
        for (int i = 0; i < 4; i++) {
          int idx = t + i * 256, n = idx >> 2, kq = (idx & 3) * 8;
          cp16_async(embB + (size_t)(code0 + n) * DC + kk + 32 + kq, sB[nxt] + n * LDT + kq);
        }
      }
      v16bf a[4];
      #pragma unroll
      for (int mf = 0; mf < 4; mf++) a[mf] = load_frag_a(sA[cur] + (mf * 16) * LDT, LDT);
      #pragma unroll
      for (int nf = 0; nf < 2; nf++) {
        v16bf bf = load_frag_b(sB[cur] + (wave * 32 + nf * 16) * LDT, LDT);
        #pragma unroll
        for (int mf = 0; mf < 4; mf++) acc[mf][nf] = wmma_bf16(a[mf], bf, acc[mf][nf]);
      }
      wait_async();
      __syncthreads();
      cur = nxt;
    }
    // fold this chunk's scores into the running argmin (dist = ||e||^2 - 2 z.e, z^2 const/row)
    int ncol = lane & 15;
    #pragma unroll
    for (int nf = 0; nf < 2; nf++) {
      int gn = code0 + wave * 32 + nf * 16 + ncol;
      float eq = esq[gn];
      #pragma unroll
      for (int mf = 0; mf < 4; mf++)
        #pragma unroll
        for (int r = 0; r < 8; r++) {
          float d = eq - 2.f * acc[mf][nf][r];
          int bi = mf * 8 + r;
          if (d < bestv[bi] || (d == bestv[bi] && gn < bestn[bi])) { bestv[bi] = d; bestn[bi] = gn; }
        }
    }
  }
  // reduce across the 16 lanes sharing each row (xor<16 stays within each half)
  #pragma unroll
  for (int bi = 0; bi < 32; bi++) {
    float v = bestv[bi]; int n = bestn[bi];
    #pragma unroll
    for (int off = 1; off < 16; off <<= 1) {
      float ov = __shfl_xor(v, off, 32);
      int   on = __shfl_xor(n, off, 32);
      if (ov < v || (ov == v && on < n)) { v = ov; n = on; }
    }
    if ((lane & 15) == 0) {
      int mf = bi >> 3, r = bi & 7;
      int m = mf * 16 + ((lane & 16) ? 8 : 0) + r;
      redV[m][wave] = v; redN[m][wave] = n;
    }
  }
  __syncthreads();
  if (t < 64) {
    float v = redV[t][0]; int n = redN[t][0];
    #pragma unroll
    for (int w = 1; w < 8; w++) {
      float ov = redV[t][w]; int on = redN[t][w];
      if (ov < v || (ov == v && on < n)) { v = ov; n = on; }
    }
    wsIdx[row0 + t] = n;
    outIdx[row0 + t] = (float)n;
  }
}

// ================= K5: gather z_q + post_proj GEMM + loss accumulation =================
__global__ void k_out(const unsigned short* __restrict__ embB,
                      const unsigned short* __restrict__ flatB,
                      const unsigned short* __restrict__ postWT,
                      const float* __restrict__ post_b, const int* __restrict__ wsIdx,
                      float* __restrict__ outQ, float* __restrict__ lossAcc) {
  __shared__ __align__(16) unsigned short sA[64 * LDT];
  __shared__ __align__(16) unsigned short sB[256 * LDT];
  __shared__ int   sIdx[64];
  __shared__ float sLoss[8];
  int row0 = blockIdx.x * 64;
  int t = threadIdx.x, wave = t >> 5, lane = t & 31;
  int wm = wave >> 1, wn = wave & 1;
  if (t < 64) sIdx[t] = wsIdx[row0 + t];
  __syncthreads();
  v8f acc[8];
  #pragma unroll
  for (int j = 0; j < 8; j++) acc[j] = vzero8();
  float lsum = 0.f;
  for (int kk = 0; kk < DC; kk += 32) {
    __syncthreads();
    // B tile: async straight to LDS
    #pragma unroll
    for (int i = 0; i < 4; i++) {
      int idx = t + i * 256, n = idx >> 2, kq = (idx & 3) * 8;
      cp16_async(postWT + (size_t)n * DC + kk + kq, sB + n * LDT + kq);
    }
    // A tile: gathered rows; values also needed in VGPRs for the loss -> synchronous
    { int r = t >> 2, kq = (t & 3) * 8;
      uint4 e4 = *(const uint4*)(embB + (size_t)sIdx[r] * DC + kk + kq);
      uint4 f4 = *(const uint4*)(flatB + (size_t)(row0 + r) * DC + kk + kq);
      *(uint4*)(sA + r * LDT + kq) = e4;
      const unsigned short* ep = (const unsigned short*)&e4;
      const unsigned short* fp = (const unsigned short*)&f4;
      #pragma unroll
      for (int i = 0; i < 8; i++) { float d = bf2f(fp[i]) - bf2f(ep[i]); lsum += d * d; }
    }
    wait_async();
    __syncthreads();
    v16bf a = load_frag_a(sA + (wm * 16) * LDT, LDT);
    #pragma unroll
    for (int j = 0; j < 8; j++) {
      v16bf bf = load_frag_b(sB + (wn * 128 + j * 16) * LDT, LDT);
      acc[j] = wmma_bf16(a, bf, acc[j]);
    }
  }
  int mrow = (lane & 16) ? 8 : 0, ncol = lane & 15;
  #pragma unroll
  for (int j = 0; j < 8; j++) {
    int gc = wn * 128 + j * 16 + ncol;
    float pb = post_b[gc];
    #pragma unroll
    for (int r = 0; r < 8; r++) {
      int gm = row0 + wm * 16 + mrow + r;
      outQ[(size_t)gm * DIMX + gc] = acc[j][r] + pb;
    }
  }
  #pragma unroll
  for (int off = 16; off; off >>= 1) lsum += __shfl_xor(lsum, off, 32);
  if (lane == 0) sLoss[wave] = lsum;
  __syncthreads();
  if (t == 0) {
    float s = 0.f;
    #pragma unroll
    for (int w = 0; w < 8; w++) s += sLoss[w];
    atomicAdd(lossAcc, s);
  }
}

// ================= K6: finalize loss =================
__global__ void k_fin(const float* __restrict__ lossAcc, float* __restrict__ outLoss) {
  if (threadIdx.x == 0 && blockIdx.x == 0)
    outLoss[0] = 1.25f * lossAcc[0] * (1.f / ((float)BN_TOK * (float)DC));
}

// ================= host launcher =================
extern "C" void kernel_launch(void* const* d_in, const int* in_sizes, int n_in,
                              void* d_out, int out_size, void* d_ws, size_t ws_size,
                              hipStream_t stream) {
  const float* x      = (const float*)d_in[0];
  const float* ln_g   = (const float*)d_in[1];
  const float* ln_b   = (const float*)d_in[2];
  const float* pre_w  = (const float*)d_in[3];
  const float* pre_b  = (const float*)d_in[4];
  const float* embed  = (const float*)d_in[5];
  const float* post_w = (const float*)d_in[6];
  const float* post_b = (const float*)d_in[7];

  char* ws = (char*)d_ws;
  unsigned short* embB   = (unsigned short*)(ws + WS_EMB);
  unsigned short* preWT  = (unsigned short*)(ws + WS_PREWT);
  unsigned short* postWT = (unsigned short*)(ws + WS_POSTWT);
  float*          esq    = (float*)(ws + WS_ESQ);
  float*          lossA  = (float*)(ws + WS_LOSS);
  int*            wsIdx  = (int*)(ws + WS_IDX);
  unsigned short* actB   = (unsigned short*)(ws + WS_ACT);
  unsigned short* flatB  = (unsigned short*)(ws + WS_FLAT);

  float* outQ    = (float*)d_out;
  float* outIdx  = outQ + (size_t)BN_TOK * DIMX;
  float* outLoss = outIdx + BN_TOK;

  k_prep  <<<2048,              256, 0, stream>>>(embed, pre_w, post_w, embB, preWT, postWT, lossA);
  k_esq   <<<KCB / 8,           256, 0, stream>>>(embed, esq);
  k_ln    <<<BN_TOK / 8,        256, 0, stream>>>(x, ln_g, ln_b, actB);
  k_gemm1 <<<(BN_TOK / 64) * 8, 256, 0, stream>>>(actB, preWT, pre_b, flatB);
  k_argmin<<<BN_TOK / 64,       256, 0, stream>>>(flatB, embB, esq, wsIdx, outIdx);
  k_out   <<<BN_TOK / 64,       256, 0, stream>>>(embB, flatB, postWT, post_b, wsIdx, outQ, lossA);
  k_fin   <<<1,                 32,  0, stream>>>(lossA, outLoss);
}
